// NaSwinAttention_12386685682341
// MI455X (gfx1250) — matrix operand order repaired and checked by reading, not compiled, not run on hardware
//
#include <hip/hip_runtime.h>

typedef __attribute__((ext_vector_type(16))) _Float16 v16h;
typedef __attribute__((ext_vector_type(8)))  _Float16 v8h;
typedef __attribute__((ext_vector_type(8)))  float    v8f;

#define N_VID    36000
#define D_MODEL  512
#define TXT_LEN  64
#define N_WIN    50
#define WIN_SIZE 720
#define SEQ      784      // 720 vid + 64 txt per window
#define KPAD     800      // 25 * 32
#define HEADS    4
#define HEAD_DIM 128

// ---------------------------------------------------------------------------
// Window geometry: wt=5, wh=9, ww=16; nt=2, nh=5, nw=5 (all exact divisions).
// Window order from the reference: w = iw*10 + ih*2 + it.
// Element order inside a window: e = (dt*9 + dh)*16 + dw.
// ---------------------------------------------------------------------------
__device__ __forceinline__ int win_token(int win, int e) {
  int it = win & 1;
  int ih = (win >> 1) % 5;
  int iw = win / 10;
  int dt = e / 144;
  int rem = e - dt * 144;
  int dh = rem >> 4;
  int dw = rem & 15;
  return (it * 5 + dt) * 3600 + (ih * 9 + dh) * 80 + (iw * 16 + dw);
}

__device__ __forceinline__ v16h load_af(const _Float16* p0, const _Float16* p1) {
  v8h a = *(const v8h*)p0;
  v8h b = *(const v8h*)p1;
  v16h r;
#pragma unroll
  for (int j = 0; j < 8; ++j) { r[j] = a[j]; r[j + 8] = b[j]; }
  return r;
}

// ---------------------------------------------------------------------------
// fp32 -> f16 copy
// ---------------------------------------------------------------------------
__global__ void cvt_kernel(const float* __restrict__ src, _Float16* __restrict__ dst,
                           long long n) {
  long long i = (long long)blockIdx.x * blockDim.x + threadIdx.x;
  if (i < n) dst[i] = (_Float16)src[i];
}

// fp32 [K x N] -> f16 [N x K] (transpose, for WMMA B operands)
__global__ void cvtT_kernel(const float* __restrict__ src, _Float16* __restrict__ dst,
                            int K, int N) {
  long long i = (long long)blockIdx.x * blockDim.x + threadIdx.x;
  long long n = (long long)K * N;
  if (i >= n) return;
  int k = (int)(i / N);
  int c = (int)(i - (long long)k * N);
  dst[(long long)c * K + k] = (_Float16)src[i];
}

// ---------------------------------------------------------------------------
// GEMM: C[M,N] = A[M,K] * B[K,N],  A f16 row-major, Bt f16 stored [N,K].
// Block = 256 thr = 8 waves. Block tile = 64 rows x 128 cols; each wave owns a
// 16-col stripe and 4 row-tiles (B-frag reused 4x). K in steps of 32,
// software-pipelined so next-step loads overlap current WMMAs.
// ---------------------------------------------------------------------------
template <bool F16OUT>
__global__ __launch_bounds__(256) void gemm_kernel(const _Float16* __restrict__ A,
                                                   const _Float16* __restrict__ Bt,
                                                   void* __restrict__ Cp,
                                                   int M, int N, int K) {
  const int wave = threadIdx.x >> 5, lane = threadIdx.x & 31;
  const int col = lane & 15, hi = lane >> 4;
  const int n0 = blockIdx.y * 128 + wave * 16;
  const int m0 = blockIdx.x * 64;

  const _Float16* brow = Bt + (size_t)(n0 + col) * K;

  // Clamp A row indices once (stores are guarded separately).
  const _Float16* arow[4];
#pragma unroll
  for (int mt = 0; mt < 4; ++mt) {
    int row = m0 + mt * 16 + col;
    if (row >= M) row = M - 1;
    arow[mt] = A + (size_t)row * K;
  }

  v8f acc[4];
#pragma unroll
  for (int mt = 0; mt < 4; ++mt)
#pragma unroll
    for (int r = 0; r < 8; ++r) acc[mt][r] = 0.f;

  v16h bf, af[4];
  bf = load_af(brow + hi * 8, brow + 16 + hi * 8);
#pragma unroll
  for (int mt = 0; mt < 4; ++mt)
    af[mt] = load_af(arow[mt] + hi * 8, arow[mt] + 16 + hi * 8);

  for (int k = 0; k < K - 32; k += 32) {
    // prefetch next step into fresh registers
    v16h bfn, afn[4];
    bfn = load_af(brow + k + 32 + hi * 8, brow + k + 48 + hi * 8);
#pragma unroll
    for (int mt = 0; mt < 4; ++mt)
      afn[mt] = load_af(arow[mt] + k + 32 + hi * 8, arow[mt] + k + 48 + hi * 8);

#pragma unroll
    for (int mt = 0; mt < 4; ++mt)
      acc[mt] = __builtin_amdgcn_wmma_f32_16x16x32_f16(false, af[mt], false, bf,
                                                       (short)0, acc[mt], false, false);
    bf = bfn;
#pragma unroll
    for (int mt = 0; mt < 4; ++mt) af[mt] = afn[mt];
  }
#pragma unroll
  for (int mt = 0; mt < 4; ++mt)
    acc[mt] = __builtin_amdgcn_wmma_f32_16x16x32_f16(false, af[mt], false, bf,
                                                     (short)0, acc[mt], false, false);

#pragma unroll
  for (int mt = 0; mt < 4; ++mt) {
#pragma unroll
    for (int r = 0; r < 8; ++r) {
      int row = m0 + mt * 16 + r + 8 * hi;
      if (row < M) {
        size_t idx = (size_t)row * N + n0 + col;
        if (F16OUT) ((_Float16*)Cp)[idx] = (_Float16)acc[mt][r];
        else        ((float*)Cp)[idx]    = acc[mt][r];
      }
    }
  }
}

// ---------------------------------------------------------------------------
// 3D RoPE in place on vid qkv (cols 0..1023 = q,k). One thread per
// (token, head, pair).  axis split: t:32, h:48, w:48 dims, interleaved pairs.
// ---------------------------------------------------------------------------
__global__ void rope_kernel(_Float16* __restrict__ qkv) {
  long long i = (long long)blockIdx.x * blockDim.x + threadIdx.x;
  if (i >= (long long)N_VID * HEADS * 64) return;
  int pair = (int)(i & 63);
  int head = (int)((i >> 6) & 3);
  int tok  = (int)(i >> 8);
  int d0 = pair * 2;

  int tt = tok / 3600;
  int hh = (tok / 80) % 45;
  int wwp = tok % 80;

  int pos, axd, dl;
  if (d0 < 32)      { pos = tt;  axd = 32; dl = d0; }
  else if (d0 < 80) { pos = hh;  axd = 48; dl = d0 - 32; }
  else              { pos = wwp; axd = 48; dl = d0 - 80; }

  double ang = (double)pos * pow(10000.0, -(double)dl / (double)axd);
  float c = (float)cos(ang);
  float s = (float)sin(ang);

  size_t base = (size_t)tok * 1536 + head * HEAD_DIM + d0;
#pragma unroll
  for (int qk = 0; qk < 2; ++qk) {  // q at +0, k at +512
    _Float16* p = qkv + base + qk * 512;
    float x0 = (float)p[0], x1 = (float)p[1];
    p[0] = (_Float16)(x0 * c - x1 * s);
    p[1] = (_Float16)(x1 * c + x0 * s);
  }
}

// ---------------------------------------------------------------------------
// Build gathered+transposed V:  Vt[win][head][d][key], key padded to 800.
// ---------------------------------------------------------------------------
__global__ void build_vt_kernel(const _Float16* __restrict__ qkv_vid,
                                const _Float16* __restrict__ qkv_txt,
                                _Float16* __restrict__ vt) {
  long long i = (long long)blockIdx.x * blockDim.x + threadIdx.x;
  if (i >= (long long)N_WIN * HEADS * HEAD_DIM * KPAD) return;
  int key  = (int)(i % KPAD);
  long long r = i / KPAD;
  int d    = (int)(r % HEAD_DIM);
  r /= HEAD_DIM;
  int head = (int)(r % HEADS);
  int win  = (int)(r / HEADS);

  _Float16 v = (_Float16)0.f;
  if (key < SEQ) {
    const _Float16* row = (key < WIN_SIZE)
        ? qkv_vid + (size_t)win_token(win, key) * 1536
        : qkv_txt + (size_t)(key - WIN_SIZE) * 1536;
    v = row[1024 + head * HEAD_DIM + d];
  }
  vt[i] = v;
}

// ---------------------------------------------------------------------------
// Flash attention per (win, head).  Block = 128 thr = 4 waves; each wave owns
// 16 queries.  Computes S^T = K*Q^T and O^T = V^T * P^T so queries live on
// the lane axis (per-lane softmax stats, no LDS transpose).
// Loads are batched ahead of the WMMAs so latency hides behind matrix ops and
// the softmax VALU work.
// ---------------------------------------------------------------------------
__global__ __launch_bounds__(128) void attn_kernel(const _Float16* __restrict__ qkv_vid,
                                                   const _Float16* __restrict__ qkv_txt,
                                                   const _Float16* __restrict__ vt,
                                                   _Float16* __restrict__ attnV,
                                                   float* __restrict__ attnT) {
  const int win  = blockIdx.x >> 2;
  const int head = blockIdx.x & 3;
  const int wave = threadIdx.x >> 5, lane = threadIdx.x & 31;
  const int col = lane & 15, hi = lane >> 4;

  const int q0   = blockIdx.y * 64 + wave * 16;
  const int qidx = q0 + col;
  const int qc   = qidx < SEQ ? qidx : SEQ - 1;  // clamp for loads

  const _Float16* qrow = (qc < WIN_SIZE)
      ? qkv_vid + (size_t)win_token(win, qc) * 1536
      : qkv_txt + (size_t)(qc - WIN_SIZE) * 1536;

  // Resident Q^T B-fragments (lane = query column, elements = d in table order)
  v16h qf[4];
#pragma unroll
  for (int c = 0; c < 4; ++c)
    qf[c] = load_af(qrow + head * HEAD_DIM + c * 32 + hi * 8, qrow + head * HEAD_DIM + c * 32 + 16 + hi * 8);

  v8f oacc[8];
#pragma unroll
  for (int t = 0; t < 8; ++t)
#pragma unroll
    for (int r = 0; r < 8; ++r) oacc[t][r] = 0.f;

  float m = -1e30f, l = 0.f;
  const float scale = 0.08838834764831845f;  // 1/sqrt(128)
  const _Float16* vbase = vt + (size_t)(win * HEADS + head) * HEAD_DIM * KPAD;

  for (int kk = 0; kk < KPAD / 32; ++kk) {
    const int kb = kk * 32;

    int k0i = kb + col;       if (k0i >= SEQ) k0i = SEQ - 1;
    int k1i = kb + 16 + col;  if (k1i >= SEQ) k1i = SEQ - 1;
    const _Float16* kr0 = (k0i < WIN_SIZE)
        ? qkv_vid + (size_t)win_token(win, k0i) * 1536
        : qkv_txt + (size_t)(k0i - WIN_SIZE) * 1536;
    const _Float16* kr1 = (k1i < WIN_SIZE)
        ? qkv_vid + (size_t)win_token(win, k1i) * 1536
        : qkv_txt + (size_t)(k1i - WIN_SIZE) * 1536;

    // ---- batch all 8 K fragments into distinct registers ----
    v16h kf[8];
#pragma unroll
    for (int c = 0; c < 4; ++c) {
      const _Float16* b0 = kr0 + 512 + head * HEAD_DIM + c * 32;
      kf[c]     = load_af(b0 + hi * 8, b0 + 16 + hi * 8);
      const _Float16* b1 = kr1 + 512 + head * HEAD_DIM + c * 32;
      kf[c + 4] = load_af(b1 + hi * 8, b1 + 16 + hi * 8);
    }

    // ---- S^T tiles: st0 (keys kb..kb+15), st1 (keys kb+16..kb+31) ----
    v8f s0, s1;
#pragma unroll
    for (int r = 0; r < 8; ++r) { s0[r] = 0.f; s1[r] = 0.f; }
#pragma unroll
    for (int c = 0; c < 4; ++c) {
      s0 = __builtin_amdgcn_wmma_f32_16x16x32_f16(false, kf[c], false, qf[c],
                                                  (short)0, s0, false, false);
      s1 = __builtin_amdgcn_wmma_f32_16x16x32_f16(false, kf[c + 4], false, qf[c],
                                                  (short)0, s1, false, false);
    }

    // ---- issue all 8 V fragments now; latency hides behind softmax ----
    v16h vfr[8];
#pragma unroll
    for (int t = 0; t < 8; ++t) {
      const _Float16* vrow = vbase + (size_t)(t * 16 + col) * KPAD + kb;
      vfr[t] = load_af(vrow + hi * 16, vrow + hi * 16 + 8);
    }

    // ---- scale + tail mask + online softmax (query = this lane) ----
    float e0[8], e1[8];
    float tmax = -1e30f;
#pragma unroll
    for (int r = 0; r < 8; ++r) {
      int key0 = kb + r + 8 * hi;
      int key1 = kb + 16 + r + 8 * hi;
      float a = (key0 < SEQ) ? s0[r] * scale : -1e30f;
      float b = (key1 < SEQ) ? s1[r] * scale : -1e30f;
      e0[r] = a; e1[r] = b;
      tmax = fmaxf(tmax, fmaxf(a, b));
    }
    tmax = fmaxf(tmax, __shfl_xor(tmax, 16, 32));  // combine with partner half
    float mnew = fmaxf(m, tmax);
    float alpha = __expf(m - mnew);
    float lsum = 0.f;
#pragma unroll
    for (int r = 0; r < 8; ++r) {
      e0[r] = __expf(e0[r] - mnew);
      e1[r] = __expf(e1[r] - mnew);
      lsum += e0[r] + e1[r];
    }
    lsum += __shfl_xor(lsum, 16, 32);
    l = l * alpha + lsum;
    m = mnew;

    // ---- assemble P^T B-fragment (32 keys x 16 queries) via lane^16 ----
    // element j -> key hi*16 + j.  Own data holds keys {r+8*hi, 16+r+8*hi}.
    v16h pf;
#pragma unroll
    for (int j = 0; j < 8; ++j) {
      float t0 = __shfl_xor(e0[j], 16, 32);
      float t1 = __shfl_xor(e1[j], 16, 32);
      pf[j]     = (_Float16)(hi ? t1 : e0[j]);
      pf[j + 8] = (_Float16)(hi ? e1[j] : t0);
    }

    // ---- O^T = alpha*O^T + V^T * P^T  (8 d-tiles of 16) ----
#pragma unroll
    for (int t = 0; t < 8; ++t) {
      v8f a = oacc[t];
#pragma unroll
      for (int r = 0; r < 8; ++r) a[r] *= alpha;
      oacc[t] = __builtin_amdgcn_wmma_f32_16x16x32_f16(false, vfr[t], false, pf,
                                                       (short)0, a, false, false);
    }
  }

  // ---- epilogue: normalize and scatter ----
  if (qidx < SEQ) {
    float inv = 1.f / l;
    if (qidx < WIN_SIZE) {
      int tok = win_token(win, qidx);
      _Float16* dst = attnV + (size_t)tok * D_MODEL + head * HEAD_DIM;
#pragma unroll
      for (int t = 0; t < 8; ++t)
#pragma unroll
        for (int r = 0; r < 8; ++r)
          dst[t * 16 + r + 8 * hi] = (_Float16)(oacc[t][r] * inv);
    } else {
      float* dst = attnT + (size_t)(qidx - WIN_SIZE) * D_MODEL + head * HEAD_DIM;
#pragma unroll
      for (int t = 0; t < 8; ++t)
#pragma unroll
        for (int r = 0; r < 8; ++r)
          atomicAdd(&dst[t * 16 + r + 8 * hi], oacc[t][r] * inv);
    }
  }
}

// mean over windows for txt rows, convert to f16
__global__ void txt_fin_kernel(const float* __restrict__ src, _Float16* __restrict__ dst) {
  int i = blockIdx.x * blockDim.x + threadIdx.x;
  if (i < TXT_LEN * D_MODEL) dst[i] = (_Float16)(src[i] * (1.f / (float)N_WIN));
}

// ---------------------------------------------------------------------------
extern "C" void kernel_launch(void* const* d_in, const int* in_sizes, int n_in,
                              void* d_out, int out_size, void* d_ws, size_t ws_size,
                              hipStream_t stream) {
  (void)in_sizes; (void)n_in; (void)out_size; (void)ws_size;

  const float* vid       = (const float*)d_in[0];
  const float* txt       = (const float*)d_in[1];
  const float* w_qkv_vid = (const float*)d_in[2];
  const float* w_qkv_txt = (const float*)d_in[3];
  const float* w_out_vid = (const float*)d_in[4];
  const float* w_out_txt = (const float*)d_in[5];
  float* out = (float*)d_out;

  char* ws = (char*)d_ws;
  size_t off = 0;
  auto carve = [&](size_t bytes) -> char* {
    char* p = ws + off;
    off = (off + bytes + 255) & ~(size_t)255;
    return p;
  };

  _Float16* vid_h  = (_Float16*)carve((size_t)N_VID * D_MODEL * 2);
  _Float16* txt_h  = (_Float16*)carve((size_t)TXT_LEN * D_MODEL * 2);
  _Float16* wqv_t  = (_Float16*)carve((size_t)1536 * 512 * 2);
  _Float16* wqt_t  = (_Float16*)carve((size_t)1536 * 512 * 2);
  _Float16* wov_t  = (_Float16*)carve((size_t)512 * 512 * 2);
  _Float16* wot_t  = (_Float16*)carve((size_t)512 * 512 * 2);
  _Float16* qkvv   = (_Float16*)carve((size_t)N_VID * 1536 * 2);
  _Float16* qkvt   = (_Float16*)carve((size_t)TXT_LEN * 1536 * 2);
  _Float16* vt     = (_Float16*)carve((size_t)N_WIN * HEADS * HEAD_DIM * KPAD * 2);
  _Float16* attnV  = (_Float16*)carve((size_t)N_VID * D_MODEL * 2);
  float*    attnTf = (float*)carve((size_t)TXT_LEN * D_MODEL * 4);
  _Float16* attnTh = (_Float16*)carve((size_t)TXT_LEN * D_MODEL * 2);

  const int B = 256;
  auto blocks = [](long long n, int b) { return (unsigned)((n + b - 1) / b); };

  // 1) f32 -> f16 conversions (+ weight transposes)
  cvt_kernel<<<blocks((long long)N_VID * 512, B), B, 0, stream>>>(vid, vid_h, (long long)N_VID * 512);
  cvt_kernel<<<blocks((long long)TXT_LEN * 512, B), B, 0, stream>>>(txt, txt_h, (long long)TXT_LEN * 512);
  cvtT_kernel<<<blocks((long long)512 * 1536, B), B, 0, stream>>>(w_qkv_vid, wqv_t, 512, 1536);
  cvtT_kernel<<<blocks((long long)512 * 1536, B), B, 0, stream>>>(w_qkv_txt, wqt_t, 512, 1536);
  cvtT_kernel<<<blocks((long long)512 * 512, B), B, 0, stream>>>(w_out_vid, wov_t, 512, 512);
  cvtT_kernel<<<blocks((long long)512 * 512, B), B, 0, stream>>>(w_out_txt, wot_t, 512, 512);

  // 2) QKV projections (f16 out)
  gemm_kernel<true><<<dim3((N_VID + 63) / 64, 1536 / 128), B, 0, stream>>>(vid_h, wqv_t, qkvv, N_VID, 1536, 512);
  gemm_kernel<true><<<dim3(1, 1536 / 128), B, 0, stream>>>(txt_h, wqt_t, qkvt, TXT_LEN, 1536, 512);

  // 3) RoPE on vid q,k (in place)
  rope_kernel<<<blocks((long long)N_VID * HEADS * 64, B), B, 0, stream>>>(qkvv);

  // 4) gathered+transposed V
  build_vt_kernel<<<blocks((long long)N_WIN * HEADS * HEAD_DIM * KPAD, B), B, 0, stream>>>(qkvv, qkvt, vt);

  // 5) attention
  hipMemsetAsync(attnTf, 0, (size_t)TXT_LEN * D_MODEL * 4, stream);
  attn_kernel<<<dim3(N_WIN * HEADS, (SEQ + 63) / 64), 128, 0, stream>>>(qkvv, qkvt, vt, attnV, attnTf);

  // 6) txt mean
  txt_fin_kernel<<<blocks(TXT_LEN * D_MODEL, B), B, 0, stream>>>(attnTf, attnTh);

  // 7) output projections (f32 out, concatenated)
  gemm_kernel<false><<<dim3((N_VID + 63) / 64, 512 / 128), B, 0, stream>>>(attnV, wov_t, out, N_VID, 512, 512);
  gemm_kernel<false><<<dim3(1, 512 / 128), B, 0, stream>>>(attnTh, wot_t, out + (size_t)N_VID * D_MODEL, TXT_LEN, 512, 512);
}
